// ONNXCompatibleAttention_12111807774748
// MI455X (gfx1250) — matrix operand order repaired and checked by reading, not compiled, not run on hardware
//
#include <hip/hip_runtime.h>

typedef unsigned short u16t;
typedef unsigned int   u32t;
typedef __attribute__((ext_vector_type(16))) __bf16 v16bf;
typedef __attribute__((ext_vector_type(8)))  float  v8f;

#define WB   2
#define WT   2048
#define WC   2048
#define WNH  16
#define WNKV 4
#define WD   128
#define WM   (WB * WT)

__device__ __forceinline__ u16t f2bf(float x) {
  u32t u = __float_as_uint(x);
  u32t r = u + 0x7FFFu + ((u >> 16) & 1u);   // round-to-nearest-even
  return (u16t)(r >> 16);
}
__device__ __forceinline__ float bf2f(u16t h) {
  return __uint_as_float(((u32t)h) << 16);
}

// A-matrix (16x32 bf16) K-pack: dword j of lane-half `hlf` holds K = kpackA, kpackA+1
__device__ __forceinline__ int kpackA(int j, int hlf) {
  return ((j & 4) ? 16 : 0) + ((j & 3) << 1) + (hlf << 3);
}
// B-matrix (32x16 bf16) K-pack: same as A: dword j of lane-half `hlf` holds K = kpackB, kpackB+1
__device__ __forceinline__ int kpackB(int j, int hlf) {
  return ((j & 4) ? 16 : 0) + ((j & 3) << 1) + (hlf << 3);
}

union FragBF {
  u32t  u[8];
  v16bf v;
};

__device__ __forceinline__ float rmax16(float x) {
  x = fmaxf(x, __shfl_xor(x, 1, 16));
  x = fmaxf(x, __shfl_xor(x, 2, 16));
  x = fmaxf(x, __shfl_xor(x, 4, 16));
  x = fmaxf(x, __shfl_xor(x, 8, 16));
  return x;
}
__device__ __forceinline__ float rsum16(float x) {
  x += __shfl_xor(x, 1, 16);
  x += __shfl_xor(x, 2, 16);
  x += __shfl_xor(x, 4, 16);
  x += __shfl_xor(x, 8, 16);
  return x;
}

// ---------------- elementwise / layout kernels ----------------

__global__ void cast_bf16_kernel(const float* __restrict__ in, u16t* __restrict__ out, int n) {
  int i = blockIdx.x * blockDim.x + threadIdx.x;
  if (i < n) out[i] = f2bf(in[i]);
}

// w[K,N] fp32  ->  wT[N,K] bf16
__global__ void transpose_bf16_kernel(const float* __restrict__ w, u16t* __restrict__ wT,
                                      int K, int N) {
  int i = blockIdx.x * blockDim.x + threadIdx.x;
  if (i >= K * N) return;
  int k = i / N, n = i % N;
  wT[(size_t)n * K + k] = f2bf(w[i]);
}

// rope: in = projection output [B*T, nheads*128] bf16, out = [B, nheads, T, 128] bf16
// `gain` folds the attention scale 1/sqrt(D) into Q (gain=1 for K)
__global__ void rope_kernel(const u16t* __restrict__ in, const float* __restrict__ cp,
                            const float* __restrict__ sp, u16t* __restrict__ out,
                            int nheads, float gain) {
  int idx = blockIdx.x * blockDim.x + threadIdx.x;
  int total = WB * WT * nheads * 64;
  if (idx >= total) return;
  int d0 = idx & 63;
  int rest = idx >> 6;
  int h = rest % nheads; rest /= nheads;
  int t = rest % WT;
  int b = rest / WT;
  int rowstride = nheads * WD;
  const u16t* base = in + (size_t)(b * WT + t) * rowstride + h * WD;
  float q0 = bf2f(base[d0]);
  float q1 = bf2f(base[d0 + 64]);
  float c0 = cp[t * WD + d0],      s0 = sp[t * WD + d0];
  float c1 = cp[t * WD + d0 + 64], s1 = sp[t * WD + d0 + 64];
  float o0 = (q0 * c0 - q1 * s0) * gain;   // rotate_half: d<64 -> -u[d+64]
  float o1 = (q1 * c1 + q0 * s1) * gain;   // d>=64 -> +u[d-64]
  u16t* ob = out + ((size_t)(b * nheads + h) * WT + t) * WD;
  ob[d0]      = f2bf(o0);
  ob[d0 + 64] = f2bf(o1);
}

// vp [B*T, 512] bf16 -> vT [B, NKV, 128, T] bf16 (transposed so P*V B-frags are contiguous)
__global__ void v_reorder_kernel(const u16t* __restrict__ vp, u16t* __restrict__ vT) {
  int idx = blockIdx.x * blockDim.x + threadIdx.x;
  int total = WB * WNKV * WT * WD;
  if (idx >= total) return;
  int d = idx & 127;
  int rest = idx >> 7;
  int h = rest % WNKV; rest /= WNKV;
  int t = rest % WT;
  int b = rest / WT;
  vT[((size_t)(b * WNKV + h) * WD + d) * WT + t] =
      vp[(size_t)(b * WT + t) * (WNKV * WD) + h * WD + d];
}

// ---------------- WMMA GEMM:  out[M,N] = A[M,K](bf16) * BT[N,K]^T(bf16) ----------------
// grid.x = M/64 (4 waves * 16 rows), grid.y = N/64, block = 128 threads

template <int BF16OUT>
__global__ void __launch_bounds__(128, 1)
gemm_wmma(const u16t* __restrict__ A, const u16t* __restrict__ BT,
          float* __restrict__ outF, u16t* __restrict__ outB,
          int N, int K) {
  const int lane = threadIdx.x & 31;
  const int wv   = threadIdx.x >> 5;
  const int n16  = lane & 15;
  const int hlf  = lane >> 4;
  const int tileM   = blockIdx.x * 4 + wv;
  const int colBase = blockIdx.y * 64;

  const u16t* arow = A + (size_t)(tileM * 16 + n16) * K;

  v8f zero = {0.f, 0.f, 0.f, 0.f, 0.f, 0.f, 0.f, 0.f};
  v8f acc[4];
#pragma unroll
  for (int t = 0; t < 4; ++t) acc[t] = zero;

  for (int kk = 0; kk < K; kk += 32) {
    FragBF a;
#pragma unroll
    for (int j = 0; j < 8; ++j)
      a.u[j] = *(const u32t*)(arow + kk + kpackA(j, hlf));
    __builtin_prefetch(arow + kk + 64, 0, 0);
#pragma unroll
    for (int t = 0; t < 4; ++t) {
      FragBF bf;
      const u16t* bcol = BT + (size_t)(colBase + t * 16 + n16) * K + kk;
#pragma unroll
      for (int j = 0; j < 8; ++j)
        bf.u[j] = *(const u32t*)(bcol + kpackB(j, hlf));
      acc[t] = __builtin_amdgcn_wmma_f32_16x16x32_bf16(
          false, a.v, false, bf.v, (short)0, acc[t], false, false);
    }
  }

#pragma unroll
  for (int t = 0; t < 4; ++t) {
#pragma unroll
    for (int r = 0; r < 8; ++r) {
      int gr = tileM * 16 + r + 8 * hlf;
      int gc = colBase + t * 16 + n16;
      float v = acc[t][r];
      if (BF16OUT) outB[(size_t)gr * N + gc] = f2bf(v);
      else         outF[(size_t)gr * N + gc] = v;
    }
  }
}

// ---------------- flash attention: 1 wave per (b, h, 16-query tile) ----------------
// Q is pre-scaled by 1/sqrt(D) at RoPE time.

__global__ void __launch_bounds__(32, 1)
attn_kernel(const u16t* __restrict__ qh, const u16t* __restrict__ kh,
            const u16t* __restrict__ vt, u16t* __restrict__ obuf) {
  __shared__ u16t psh[16 * 32];
  const int lane = threadIdx.x & 31;
  const int n16  = lane & 15;
  const int hlf  = lane >> 4;
  const int qt = blockIdx.x;
  const int h  = blockIdx.y;
  const int b  = blockIdx.z;
  const int hkv = h >> 2;                       // N_REP = 4

  // Q fragments: 16x128 as 4 A-frags of 16x32
  FragBF qf[4];
  const u16t* qrow = qh + ((size_t)(b * WNH + h) * WT + qt * 16 + n16) * WD;
#pragma unroll
  for (int f = 0; f < 4; ++f)
#pragma unroll
    for (int j = 0; j < 8; ++j)
      qf[f].u[j] = *(const u32t*)(qrow + f * 32 + kpackA(j, hlf));

  v8f zero = {0.f, 0.f, 0.f, 0.f, 0.f, 0.f, 0.f, 0.f};
  v8f acc[8];
#pragma unroll
  for (int dt = 0; dt < 8; ++dt) acc[dt] = zero;
  float mr[8], lr[8];
#pragma unroll
  for (int r = 0; r < 8; ++r) { mr[r] = -1e30f; lr[r] = 0.f; }

  const u16t* kbasep = kh + (size_t)(b * WNKV + hkv) * WT * WD;   // [T, D]
  const u16t* vbasep = vt + (size_t)(b * WNKV + hkv) * WD * WT;   // [D, T]
  const int cmax = (qt * 16 + 15) >> 5;                           // causal bound

  for (int c = 0; c <= cmax; ++c) {
    const int k0 = c * 32;
    // S = Q * K^T for 32 keys: two 16x16 tiles, K-dim = 128 (4 steps).
    // One B-fragment temp live at a time to limit pressure.
    v8f s0 = zero, s1 = zero;
#pragma unroll
    for (int ks = 0; ks < 4; ++ks) {
      FragBF bfr;
      const u16t* kc = kbasep + (size_t)(k0 + n16) * WD + ks * 32;
#pragma unroll
      for (int j = 0; j < 8; ++j)
        bfr.u[j] = *(const u32t*)(kc + kpackB(j, hlf));
      s0 = __builtin_amdgcn_wmma_f32_16x16x32_bf16(false, qf[ks].v, false, bfr.v,
                                                   (short)0, s0, false, false);
    }
#pragma unroll
    for (int ks = 0; ks < 4; ++ks) {
      FragBF bfr;
      const u16t* kc = kbasep + (size_t)(k0 + 16 + n16) * WD + ks * 32;
#pragma unroll
      for (int j = 0; j < 8; ++j)
        bfr.u[j] = *(const u32t*)(kc + kpackB(j, hlf));
      s1 = __builtin_amdgcn_wmma_f32_16x16x32_bf16(false, qf[ks].v, false, bfr.v,
                                                   (short)0, s1, false, false);
    }

    // online softmax (rows r+8*hlf, cols across the 16 lanes of each half);
    // P converted to bf16 and stored to LDS immediately (no p-arrays kept live)
    __syncthreads();
    float al[8];
#pragma unroll
    for (int r = 0; r < 8; ++r) {
      const int rq = qt * 16 + r + 8 * hlf;
      float v0 = s0[r];
      float v1 = s1[r];
      if (k0 + n16 > rq)      v0 = -1e30f;
      if (k0 + 16 + n16 > rq) v1 = -1e30f;
      float m  = rmax16(fmaxf(v0, v1));
      float mn = fmaxf(mr[r], m);
      float a  = __expf(mr[r] - mn);
      float p0 = __expf(v0 - mn);
      float p1 = __expf(v1 - mn);
      float rs = rsum16(p0 + p1);
      lr[r] = lr[r] * a + rs;
      mr[r] = mn;
      al[r] = a;
      psh[(r + 8 * hlf) * 32 + n16]      = f2bf(p0);
      psh[(r + 8 * hlf) * 32 + 16 + n16] = f2bf(p1);
    }
#pragma unroll
    for (int dt = 0; dt < 8; ++dt)
#pragma unroll
      for (int r = 0; r < 8; ++r) acc[dt][r] *= al[r];
    __syncthreads();

    FragBF pf;
#pragma unroll
    for (int j = 0; j < 8; ++j)
      pf.u[j] = *(const u32t*)&psh[n16 * 32 + kpackA(j, hlf)];

    // O += P (16x32) * V (32x128): 8 d-tiles
#pragma unroll
    for (int dt = 0; dt < 8; ++dt) {
      FragBF vf;
      const u16t* vc = vbasep + (size_t)(dt * 16 + n16) * WT + k0;
#pragma unroll
      for (int j = 0; j < 8; ++j)
        vf.u[j] = *(const u32t*)(vc + kpackB(j, hlf));
      acc[dt] = __builtin_amdgcn_wmma_f32_16x16x32_bf16(
          false, pf.v, false, vf.v, (short)0, acc[dt], false, false);
    }
  }

  // normalize and write O in [B, T, H*D] layout (bf16) for the output GEMM
  float inv[8];
#pragma unroll
  for (int r = 0; r < 8; ++r) inv[r] = 1.f / lr[r];
#pragma unroll
  for (int dt = 0; dt < 8; ++dt)
#pragma unroll
    for (int r = 0; r < 8; ++r) {
      const int t   = qt * 16 + r + 8 * hlf;
      const int col = h * WD + dt * 16 + n16;
      obuf[(size_t)(b * WT + t) * WC + col] = f2bf(acc[dt][r] * inv[r]);
    }
}

// ---------------- launch ----------------

extern "C" void kernel_launch(void* const* d_in, const int* in_sizes, int n_in,
                              void* d_out, int out_size, void* d_ws, size_t ws_size,
                              hipStream_t stream) {
  const float* x        = (const float*)d_in[0];
  const float* rope_cos = (const float*)d_in[1];
  const float* rope_sin = (const float*)d_in[2];
  const float* wq       = (const float*)d_in[3];
  const float* wk       = (const float*)d_in[4];
  const float* wv       = (const float*)d_in[5];
  const float* wo       = (const float*)d_in[6];
  float* out = (float*)d_out;

  char* ws = (char*)d_ws;
  size_t off = 0;
  auto alloc = [&](size_t bytes) -> char* {
    char* p = ws + off;
    off += (bytes + 255) & ~(size_t)255;
    return p;
  };
  u16t* xh   = (u16t*)alloc((size_t)WM * WC * 2);
  u16t* wqT  = (u16t*)alloc((size_t)WC * WC * 2);
  u16t* wkT  = (u16t*)alloc((size_t)512 * WC * 2);
  u16t* wvT  = (u16t*)alloc((size_t)512 * WC * 2);
  u16t* woT  = (u16t*)alloc((size_t)WC * WC * 2);
  u16t* qp   = (u16t*)alloc((size_t)WM * WC * 2);
  u16t* kp   = (u16t*)alloc((size_t)WM * 512 * 2);
  u16t* vp   = (u16t*)alloc((size_t)WM * 512 * 2);
  u16t* qhat = (u16t*)alloc((size_t)WB * WNH * WT * WD * 2);
  u16t* khat = (u16t*)alloc((size_t)WB * WNKV * WT * WD * 2);
  u16t* vT   = (u16t*)alloc((size_t)WB * WNKV * WD * WT * 2);
  u16t* obuf = (u16t*)alloc((size_t)WM * WC * 2);

  {
    int n = WM * WC;
    cast_bf16_kernel<<<(n + 255) / 256, 256, 0, stream>>>(x, xh, n);
  }
  transpose_bf16_kernel<<<(WC * WC + 255) / 256, 256, 0, stream>>>(wq, wqT, WC, WC);
  transpose_bf16_kernel<<<(WC * 512 + 255) / 256, 256, 0, stream>>>(wk, wkT, WC, 512);
  transpose_bf16_kernel<<<(WC * 512 + 255) / 256, 256, 0, stream>>>(wv, wvT, WC, 512);
  transpose_bf16_kernel<<<(WC * WC + 255) / 256, 256, 0, stream>>>(wo, woT, WC, WC);

  dim3 blk(128);
  gemm_wmma<1><<<dim3(WM / 64, WC / 64), blk, 0, stream>>>(xh, wqT, nullptr, qp, WC, WC);
  gemm_wmma<1><<<dim3(WM / 64, 512 / 64), blk, 0, stream>>>(xh, wkT, nullptr, kp, 512, WC);
  gemm_wmma<1><<<dim3(WM / 64, 512 / 64), blk, 0, stream>>>(xh, wvT, nullptr, vp, 512, WC);

  const float qscale = 0.08838834764831845f;   // 1/sqrt(128)
  rope_kernel<<<(WB * WT * WNH * 64 + 255) / 256, 256, 0, stream>>>(qp, rope_cos, rope_sin,
                                                                    qhat, WNH, qscale);
  rope_kernel<<<(WB * WT * WNKV * 64 + 255) / 256, 256, 0, stream>>>(kp, rope_cos, rope_sin,
                                                                     khat, WNKV, 1.0f);
  v_reorder_kernel<<<(WB * WNKV * WT * WD + 255) / 256, 256, 0, stream>>>(vp, vT);

  attn_kernel<<<dim3(WT / 16, WNH, WB), 32, 0, stream>>>(qhat, khat, vT, obuf);

  gemm_wmma<0><<<dim3(WM / 64, WC / 64), blk, 0, stream>>>(obuf, woT, out, nullptr, WC, WC);
}
